// Transformer_91311004713226
// MI455X (gfx1250) — matrix operand order repaired and checked by reading, not compiled, not run on hardware
//
#include <hip/hip_runtime.h>

// ---------------------------------------------------------------------------
// CDNA5 (gfx1250) Transformer forward. wave32, WMMA bf16 16x16x32 everywhere.
// ---------------------------------------------------------------------------

typedef __bf16 bf16_t;
typedef bf16_t v16bf __attribute__((ext_vector_type(16)));
typedef bf16_t v2bf  __attribute__((ext_vector_type(2)));
typedef float  v8f   __attribute__((ext_vector_type(8)));
typedef float  f4    __attribute__((ext_vector_type(4)));
typedef unsigned int v8u __attribute__((ext_vector_type(8)));

#define D_MODEL 512
#define NHEAD   8
#define DKH     64
#define NEGF   (-3.402823466e38f)

// pack two f32 -> two bf16 via native converts (hardware RNE)
__device__ __forceinline__ unsigned int pk2(float a, float b) {
  v2bf v;
  v.x = (bf16_t)a;
  v.y = (bf16_t)b;
  return __builtin_bit_cast(unsigned int, v);
}

__device__ __forceinline__ v8f wmma_bf16(const v8u& a, const v8u& b, v8f c) {
  return __builtin_amdgcn_wmma_f32_16x16x32_bf16(
      false, __builtin_bit_cast(v16bf, a),
      false, __builtin_bit_cast(v16bf, b),
      (short)0, c, false, false);
}

// T5 relative position bucket (NUM_BUCKETS=32, MAX_DIST=512)
__device__ __forceinline__ int rel_bucket(int q, int k) {
  int n = q - k; if (n < 0) n = 0;
  if (n < 16) return n;
  int v = 16 + (int)(__logf((float)n * 0.0625f) * (16.0f / 3.4657359028f)); // /ln(32)
  return v < 31 ? v : 31;
}

// ---------------------------------------------------------------------------
// Embedding * sqrt(D) + sinusoidal positional encoding (computed analytically)
// ---------------------------------------------------------------------------
__global__ void __launch_bounds__(256)
embed_kernel(const int* __restrict__ tok, const float* __restrict__ emb,
             float* __restrict__ out, int S) {
  int i = blockIdx.x;            // b*S + s
  int s = i % S;
  int t = tok[i];
  for (int c = threadIdx.x; c < D_MODEL; c += 256) {
    int j2 = c & ~1;             // 2*floor(c/2)
    float div = __expf((float)j2 * (-9.210340372f / (float)D_MODEL)); // ln(1e4)
    float arg = (float)s * div;
    float pe  = (c & 1) ? __cosf(arg) : __sinf(arg);
    out[(size_t)i * D_MODEL + c] =
        emb[(size_t)t * D_MODEL + c] * 22.62741699797f + pe; // sqrt(512)
  }
}

// ---------------------------------------------------------------------------
// GEMM: C[M,N] = act(A[M,K] @ W[K,N] + bias). 64x128 block tile, 8 waves,
// 4 accumulators per wave (A-fragment reused across 4 WMMAs per K-step).
// Staging issues all 6 b128 loads before converting (loads stay in flight);
// A temporal, W non-temporal (weights are a single-use stream past L2).
// qkv_scatter=1 rewrites output to (B,H,Srow,64) head-major layout.
// ---------------------------------------------------------------------------
__global__ void __launch_bounds__(256)
gemm_kernel(const float* __restrict__ A, const float* __restrict__ W,
            const float* __restrict__ bias, float* __restrict__ C,
            int M, int N, int K, int relu, int qkv_scatter, int Sdim) {
  __shared__ unsigned int lA[64 * 17];    // 64 rows  x 16 k-pairs (+pad)
  __shared__ unsigned int lB[128 * 17];   // 128 cols x 16 k-pairs (+pad)

  const int tid  = threadIdx.x;
  const int lane = tid & 31;
  const int wave = tid >> 5;
  const int m0 = blockIdx.y * 64;
  const int n0 = blockIdx.x * 128;
  const int wm = wave & 3;   // 16-row tile
  const int wn = wave >> 2;  // 64-col half

  // staging coordinates (fixed per thread)
  const int rowA = tid >> 3;            // 0..31 (pass1: +32)
  const int pqA  = (tid & 7) * 2;       // pair index within row
  const int kkB  = tid >> 5;            // 0..7  (pass1: +8)
  const int c4B  = (tid & 31) * 4;      // column quad

  v8f acc[4];
  #pragma unroll
  for (int t = 0; t < 4; ++t) acc[t] = (v8f){};

  for (int kt = 0; kt < K; kt += 32) {
    // ---- issue ALL global loads first (6 x b128 in flight) ----
    const f4 a0 = *(const f4*)(A + (size_t)(m0 + rowA) * K + kt + pqA * 2);
    const f4 a1 = *(const f4*)(A + (size_t)(m0 + rowA + 32) * K + kt + pqA * 2);
    const float* wp0 = W + (size_t)(kt + kkB * 2) * N + n0 + c4B;
    const float* wp1 = W + (size_t)(kt + (kkB + 8) * 2) * N + n0 + c4B;
    const f4 w00 = __builtin_nontemporal_load((const f4*)wp0);
    const f4 w01 = __builtin_nontemporal_load((const f4*)(wp0 + N));
    const f4 w10 = __builtin_nontemporal_load((const f4*)wp1);
    const f4 w11 = __builtin_nontemporal_load((const f4*)(wp1 + N));
    // warm next K-tile while this one computes (global_prefetch_b8)
    if (kt + 32 < K) {
      __builtin_prefetch(A + (size_t)(m0 + (tid & 63)) * K + kt + 32 + ((tid >> 6) << 4), 0, 1);
      __builtin_prefetch(W + (size_t)(kt + 32 + ((tid >> 3) & 31)) * N + n0 + ((tid & 7) << 4), 0, 1);
    }
    // ---- convert + store to LDS ----
    lA[rowA * 17 + pqA]            = pk2(a0.x, a0.y);
    lA[rowA * 17 + pqA + 1]        = pk2(a0.z, a0.w);
    lA[(rowA + 32) * 17 + pqA]     = pk2(a1.x, a1.y);
    lA[(rowA + 32) * 17 + pqA + 1] = pk2(a1.z, a1.w);
    lB[(c4B + 0) * 17 + kkB]       = pk2(w00.x, w01.x);
    lB[(c4B + 1) * 17 + kkB]       = pk2(w00.y, w01.y);
    lB[(c4B + 2) * 17 + kkB]       = pk2(w00.z, w01.z);
    lB[(c4B + 3) * 17 + kkB]       = pk2(w00.w, w01.w);
    lB[(c4B + 0) * 17 + kkB + 8]   = pk2(w10.x, w11.x);
    lB[(c4B + 1) * 17 + kkB + 8]   = pk2(w10.y, w11.y);
    lB[(c4B + 2) * 17 + kkB + 8]   = pk2(w10.z, w11.z);
    lB[(c4B + 3) * 17 + kkB + 8]   = pk2(w10.w, w11.w);
    __syncthreads();

    // A fragment: row = wm*16 + lane&15, K-half by lane>>4
    const int arow = wm * 16 + (lane & 15);
    const int kh2  = (lane >> 4) * 8;
    v8u af;
    #pragma unroll
    for (int r = 0; r < 4; ++r) {
      af[r]     = lA[arow * 17 + kh2 + r];
      af[r + 4] = lA[arow * 17 + 4 + kh2 + r];
    }
    // 4 B fragments, A reused across 4 WMMAs
    const int kb2 = (lane >> 4) * 8;
    #pragma unroll
    for (int t = 0; t < 4; ++t) {
      const int bc = wn * 64 + t * 16 + (lane & 15);
      v8u bf;
      #pragma unroll
      for (int r = 0; r < 8; ++r) bf[r] = lB[bc * 17 + kb2 + r];
      acc[t] = wmma_bf16(af, bf, acc[t]);
    }
    __syncthreads();
  }

  // epilogue: C/D layout -> row = (lane>>4)*8 + r, col = lane&15
  const int rbase = m0 + wm * 16 + (lane >> 4) * 8;
  #pragma unroll
  for (int t = 0; t < 4; ++t) {
    int col = n0 + wn * 64 + t * 16 + (lane & 15);
    float bv = bias ? bias[col] : 0.0f;
    #pragma unroll
    for (int r = 0; r < 8; ++r) {
      float v = acc[t][r] + bv;
      if (relu) v = fmaxf(v, 0.0f);
      int row = rbase + r;
      if (qkv_scatter) {
        int b = row / Sdim, s = row - b * Sdim;
        int h = col >> 6, dk = col & 63;
        C[(((size_t)(b * NHEAD + h) * Sdim) + s) * DKH + dk] = v;
      } else {
        C[(size_t)row * N + col] = v;
      }
    }
  }
}

// ---------------------------------------------------------------------------
// Attention: one (b,h) x 16-query tile per workgroup, 8 waves.
// scores = (Q*0.125) @ K^T (+ rpe bias) (+ causal) -> softmax in LDS -> P @ V.
// Q/K/V layout: (B,H,Skv,64). Out layout: (B,Sq,512) at head column offset.
// ---------------------------------------------------------------------------
__global__ void __launch_bounds__(256)
attn_kernel(const float* __restrict__ Q, const float* __restrict__ Kt,
            const float* __restrict__ V, const float* __restrict__ rpe,
            float* __restrict__ Out, int Sq, int Skv, int causal) {
  __shared__ float        sc[16 * 520];      // score rows (pad 520)
  __shared__ unsigned int lQ[16 * 33];       // scaled Q bf16 pairs
  __shared__ float        rinv[16];
  __shared__ float        lO[2][16 * 64];    // K-half partials of P@V

  const int tid  = threadIdx.x;
  const int lane = tid & 31;
  const int wave = tid >> 5;
  const int bh = blockIdx.y;
  const int h  = bh & (NHEAD - 1);
  const int b  = bh >> 3;
  const int q0 = blockIdx.x * 16;

  const float* Qp = Q  + ((size_t)bh * Sq + q0) * DKH;
  const float* Kp = Kt + (size_t)bh * Skv * DKH;
  const float* Vp = V  + (size_t)bh * Skv * DKH;

  // load Q tile (fold 1/sqrt(64) scale): 16x32 pairs = 256 quads, one pass
  {
    const int row = tid >> 4;
    const int pq  = (tid & 15) * 2;
    const f4 v = *(const f4*)(Qp + row * DKH + pq * 2);
    lQ[row * 33 + pq]     = pk2(v.x * 0.125f, v.y * 0.125f);
    lQ[row * 33 + pq + 1] = pk2(v.z * 0.125f, v.w * 0.125f);
  }
  __syncthreads();

  // A fragments (d 0..31 and 32..63), shared by all column tiles of this wave
  const int arow = lane & 15;
  const int kh2  = (lane >> 4) * 8;
  v8u aq0, aq1;
  #pragma unroll
  for (int r = 0; r < 4; ++r) {
    aq0[r]     = lQ[arow * 33 + kh2 + r];
    aq0[r + 4] = lQ[arow * 33 + 4 + kh2 + r];
    aq1[r]     = lQ[arow * 33 + 16 + kh2 + r];
    aq1[r + 4] = lQ[arow * 33 + 20 + kh2 + r];
  }

  // ---- scores: 32 column tiles striped over 8 waves
  for (int ct = wave; ct < (Skv >> 4); ct += 8) {
    const int key = ct * 16 + (lane & 15);
    const int kb  = (lane >> 4) * 16;
    const float* kp = Kp + (size_t)key * DKH + kb;
    v8u b0, b1;
    #pragma unroll
    for (int r = 0; r < 8; ++r) {
      b0[r] = pk2(kp[2 * r],      kp[2 * r + 1]);        // d = kb+2r
      b1[r] = pk2(kp[32 + 2 * r], kp[32 + 2 * r + 1]);   // d = 32+kb+2r
    }
    v8f a = (v8f){};
    a = wmma_bf16(aq0, b0, a);
    a = wmma_bf16(aq1, b1, a);
    const int rb = (lane >> 4) * 8;
    const int cc = ct * 16 + (lane & 15);
    #pragma unroll
    for (int r = 0; r < 8; ++r) {
      int qg = q0 + rb + r;
      float v = a[r];
      if (rpe) v += rpe[rel_bucket(qg, cc) * NHEAD + h];
      if (causal && cc > qg) v = NEGF;
      sc[(rb + r) * 520 + cc] = v;
    }
  }
  __syncthreads();

  // ---- softmax: 16 threads per row, shuffle-tree reductions (wave32-safe,
  //      xor masks <= 8 stay inside each 16-lane row group)
  {
    const int row = tid >> 4, j = tid & 15;
    float mx = NEGF;
    for (int c = j; c < Skv; c += 16) mx = fmaxf(mx, sc[row * 520 + c]);
    #pragma unroll
    for (int off = 8; off > 0; off >>= 1) mx = fmaxf(mx, __shfl_xor(mx, off));
    float sum = 0.0f;
    for (int c = j; c < Skv; c += 16) {
      float e = __expf(sc[row * 520 + c] - mx);
      sc[row * 520 + c] = e;
      sum += e;
    }
    #pragma unroll
    for (int off = 8; off > 0; off >>= 1) sum += __shfl_xor(sum, off);
    if (j == 0) rinv[row] = 1.0f / sum;
  }
  __syncthreads();

  // ---- P @ V : waves split as (4 d-tiles) x (2 K-halves)
  const int dt = wave & 3, ks = wave >> 2;
  const int half = Skv >> 1;
  v8f oacc = (v8f){};
  for (int kk = ks * half; kk < ks * half + half; kk += 32) {
    const int row = lane & 15;
    const int khh = (lane >> 4) * 16;
    const float inv = rinv[row];
    v8u pa;
    #pragma unroll
    for (int r = 0; r < 4; ++r) {
      int c0 = kk + khh + 2 * r;
      int c1 = kk + 8 + khh + 2 * r;
      pa[r]     = pk2(sc[row * 520 + c0] * inv, sc[row * 520 + c0 + 1] * inv);
      pa[r + 4] = pk2(sc[row * 520 + c1] * inv, sc[row * 520 + c1 + 1] * inv);
    }
    const int col = dt * 16 + (lane & 15);
    const int kb  = (lane >> 4) * 16;
    v8u vb;
    #pragma unroll
    for (int r = 0; r < 8; ++r) {
      const float* vp = Vp + (size_t)(kk + kb + 2 * r) * DKH + col;
      vb[r] = pk2(vp[0], vp[DKH]);
    }
    oacc = wmma_bf16(pa, vb, oacc);
  }
  {
    const int rb  = (lane >> 4) * 8;
    const int col = dt * 16 + (lane & 15);
    #pragma unroll
    for (int r = 0; r < 8; ++r) lO[ks][(rb + r) * 64 + col] = oacc[r];
  }
  __syncthreads();

  for (int p = tid; p < 16 * 64; p += 256) {
    int row = p >> 6, col = p & 63;
    float v = lO[0][p] + lO[1][p];
    Out[((size_t)(b * Sq + q0 + row)) * D_MODEL + h * DKH + col] = v;
  }
}

// ---------------------------------------------------------------------------
// Fused residual add + LayerNorm over D=512. One row per block, in place.
// ---------------------------------------------------------------------------
__global__ void __launch_bounds__(256)
addln_kernel(float* __restrict__ X, const float* __restrict__ R,
             const float* __restrict__ g, const float* __restrict__ bb) {
  __shared__ float red1[8], red2[8];
  const int row = blockIdx.x;
  const int tid = threadIdx.x;
  float* xp = X + (size_t)row * D_MODEL;
  const float* rp = R + (size_t)row * D_MODEL;

  float v0 = xp[tid] + rp[tid];
  float v1 = xp[tid + 256] + rp[tid + 256];

  float s = v0 + v1;
  #pragma unroll
  for (int off = 16; off > 0; off >>= 1) s += __shfl_xor(s, off);
  if ((tid & 31) == 0) red1[tid >> 5] = s;
  __syncthreads();
  float tot = 0.0f;
  #pragma unroll
  for (int r = 0; r < 8; ++r) tot += red1[r];
  const float m = tot * (1.0f / (float)D_MODEL);

  float d0 = v0 - m, d1 = v1 - m;
  float vs = d0 * d0 + d1 * d1;
  #pragma unroll
  for (int off = 16; off > 0; off >>= 1) vs += __shfl_xor(vs, off);
  if ((tid & 31) == 0) red2[tid >> 5] = vs;
  __syncthreads();
  float vtot = 0.0f;
  #pragma unroll
  for (int r = 0; r < 8; ++r) vtot += red2[r];
  const float rs = rsqrtf(vtot * (1.0f / (float)D_MODEL) + 1e-5f);

  xp[tid]       = d0 * rs * g[tid]       + bb[tid];
  xp[tid + 256] = d1 * rs * g[tid + 256] + bb[tid + 256];
}

// ---------------------------------------------------------------------------
// Host-side orchestration
// ---------------------------------------------------------------------------
enum {
  IN_SRC = 0, IN_TGT, IN_SRCMASK, IN_TGTMASK, IN_SKPM, IN_TKPM, IN_MKPM,
  DEC_B1, DEC_B2, DEC_CBK, DEC_CBO, DEC_CBQ, DEC_CBV,
  DEC_CWK, DEC_CWO, DEC_CWQ, DEC_CWV,
  DEC_LN1B, DEC_LN1G, DEC_LN2B, DEC_LN2G, DEC_LN3B, DEC_LN3G,
  DEC_SBK, DEC_SBO, DEC_SBQ, DEC_SBV, DEC_SRPE,
  DEC_SWK, DEC_SWO, DEC_SWQ, DEC_SWV, DEC_W1, DEC_W2,
  ENC_B1, ENC_B2, ENC_BK, ENC_BO, ENC_BQ, ENC_BV,
  ENC_LN1B, ENC_LN1G, ENC_LN2B, ENC_LN2G, ENC_RPE,
  ENC_W1, ENC_W2, ENC_WK, ENC_WO, ENC_WQ, ENC_WV,
  GEN_B, GEN_W, SRC_EMB, TGT_EMB
};

extern "C" void kernel_launch(void* const* d_in, const int* in_sizes, int n_in,
                              void* d_out, int out_size, void* d_ws, size_t ws_size,
                              hipStream_t stream) {
  (void)in_sizes; (void)out_size; (void)ws_size;
  const int B = 4, S = 512, T = 512, L = 6, DFF = 2048, V = 32000;
  const int M = B * S;                       // == B*T == 2048

  const float* P[64];
  for (int i = 0; i < n_in && i < 64; ++i) P[i] = (const float*)d_in[i];
  const int* src = (const int*)d_in[IN_SRC];
  const int* tgt = (const int*)d_in[IN_TGT];

  float* ws = (float*)d_ws;
  const size_t MS = (size_t)M * D_MODEL;     // 1M floats
  float* X  = ws;            // encoder activations -> becomes memory
  float* Y  = ws + 1 * MS;   // decoder activations
  float* Qb = ws + 2 * MS;
  float* Kb = ws + 3 * MS;
  float* Vb = ws + 4 * MS;
  float* Ab = ws + 5 * MS;   // attention context (B,S,D)
  float* T2 = ws + 6 * MS;   // projection / FF2 output
  float* T1 = ws + 7 * MS;   // FF hidden (B*S, 2048) = 4M floats

  auto gemm = [&](const float* A, const float* W, const float* bias, float* C,
                  int m, int n, int k, int relu, int scat) {
    gemm_kernel<<<dim3(n / 128, m / 64), 256, 0, stream>>>(A, W, bias, C,
                                                           m, n, k, relu, scat, S);
  };
  auto attn = [&](const float* q, const float* k, const float* v,
                  const float* rpe, float* o, int causal) {
    attn_kernel<<<dim3(S / 16, B * NHEAD), 256, 0, stream>>>(q, k, v, rpe, o,
                                                             S, S, causal);
  };
  auto addln = [&](float* x, const float* r, const float* g, const float* bb) {
    addln_kernel<<<dim3(M), 256, 0, stream>>>(x, r, g, bb);
  };

  const size_t DD = (size_t)D_MODEL * D_MODEL;
  const size_t DF = (size_t)D_MODEL * DFF;

  // ---------------- encoder ----------------
  embed_kernel<<<dim3(M), 256, 0, stream>>>(src, P[SRC_EMB], X, S);
  for (int l = 0; l < L; ++l) {
    gemm(X, P[ENC_WQ] + l * DD, P[ENC_BQ] + l * D_MODEL, Qb, M, D_MODEL, D_MODEL, 0, 1);
    gemm(X, P[ENC_WK] + l * DD, P[ENC_BK] + l * D_MODEL, Kb, M, D_MODEL, D_MODEL, 0, 1);
    gemm(X, P[ENC_WV] + l * DD, P[ENC_BV] + l * D_MODEL, Vb, M, D_MODEL, D_MODEL, 0, 1);
    attn(Qb, Kb, Vb, P[ENC_RPE] + l * 32 * NHEAD, Ab, /*causal=*/0);
    gemm(Ab, P[ENC_WO] + l * DD, P[ENC_BO] + l * D_MODEL, T2, M, D_MODEL, D_MODEL, 0, 0);
    addln(X, T2, P[ENC_LN1G] + l * D_MODEL, P[ENC_LN1B] + l * D_MODEL);
    gemm(X, P[ENC_W1] + l * DF, P[ENC_B1] + l * DFF, T1, M, DFF, D_MODEL, /*relu=*/1, 0);
    gemm(T1, P[ENC_W2] + l * DF, P[ENC_B2] + l * D_MODEL, T2, M, D_MODEL, DFF, 0, 0);
    addln(X, T2, P[ENC_LN2G] + l * D_MODEL, P[ENC_LN2B] + l * D_MODEL);
  }
  // X now holds encoder memory.

  // ---------------- decoder ----------------
  embed_kernel<<<dim3(M), 256, 0, stream>>>(tgt, P[TGT_EMB], Y, T);
  for (int l = 0; l < L; ++l) {
    // masked self-attention (+ relative position bias)
    gemm(Y, P[DEC_SWQ] + l * DD, P[DEC_SBQ] + l * D_MODEL, Qb, M, D_MODEL, D_MODEL, 0, 1);
    gemm(Y, P[DEC_SWK] + l * DD, P[DEC_SBK] + l * D_MODEL, Kb, M, D_MODEL, D_MODEL, 0, 1);
    gemm(Y, P[DEC_SWV] + l * DD, P[DEC_SBV] + l * D_MODEL, Vb, M, D_MODEL, D_MODEL, 0, 1);
    attn(Qb, Kb, Vb, P[DEC_SRPE] + l * 32 * NHEAD, Ab, /*causal=*/1);
    gemm(Ab, P[DEC_SWO] + l * DD, P[DEC_SBO] + l * D_MODEL, T2, M, D_MODEL, D_MODEL, 0, 0);
    addln(Y, T2, P[DEC_LN1G] + l * D_MODEL, P[DEC_LN1B] + l * D_MODEL);
    // cross-attention against encoder memory
    gemm(Y, P[DEC_CWQ] + l * DD, P[DEC_CBQ] + l * D_MODEL, Qb, M, D_MODEL, D_MODEL, 0, 1);
    gemm(X, P[DEC_CWK] + l * DD, P[DEC_CBK] + l * D_MODEL, Kb, M, D_MODEL, D_MODEL, 0, 1);
    gemm(X, P[DEC_CWV] + l * DD, P[DEC_CBV] + l * D_MODEL, Vb, M, D_MODEL, D_MODEL, 0, 1);
    attn(Qb, Kb, Vb, nullptr, Ab, /*causal=*/0);
    gemm(Ab, P[DEC_CWO] + l * DD, P[DEC_CBO] + l * D_MODEL, T2, M, D_MODEL, D_MODEL, 0, 0);
    addln(Y, T2, P[DEC_LN2G] + l * D_MODEL, P[DEC_LN2B] + l * D_MODEL);
    // feed-forward
    gemm(Y, P[DEC_W1] + l * DF, P[DEC_B1] + l * DFF, T1, M, DFF, D_MODEL, /*relu=*/1, 0);
    gemm(T1, P[DEC_W2] + l * DF, P[DEC_B2] + l * D_MODEL, T2, M, D_MODEL, DFF, 0, 0);
    addln(Y, T2, P[DEC_LN3G] + l * D_MODEL, P[DEC_LN3B] + l * D_MODEL);
  }

  // ---------------- generator ----------------
  gemm(Y, P[GEN_W], P[GEN_B], (float*)d_out, M, V, D_MODEL, 0, 0);
}
